// SymmetricContraction_171798692582
// MI455X (gfx1250) — matrix operand order repaired and checked by reading, not compiled, not run on hardware
//
#include <hip/hip_runtime.h>

typedef __attribute__((ext_vector_type(16))) __bf16        v16bf;
typedef __attribute__((ext_vector_type(8)))  float         v8f;
typedef __attribute__((ext_vector_type(8)))  unsigned int  v8u;

#define BN    1024
#define CN    256
#define LN    16
#define EN    10
#define K3N   23
#define K2N   4
#define KKUSE 372          // 368 (U3) + 4 (U2)
#define KKPAD 384          // padded to 12 * 32
#define NKS   12           // K-steps of 32
#define RM    64           // rows (c values) per workgroup
#define NTHR  64           // 2 waves; each wave owns 2 M-tiles (B reuse x2)

// round-to-nearest-even f32 -> bf16 (as raw u16)
__device__ __forceinline__ unsigned short f2bf(float f) {
    unsigned int u = __builtin_bit_cast(unsigned int, f);
    u += 0x7FFFu + ((u >> 16) & 1u);
    return (unsigned short)(u >> 16);
}

// element of the extended [N=256, K=384] "U3T" matrix (row n, col kk)
__device__ __forceinline__ float u3t_elem(const float* __restrict__ U3,
                                          const float* __restrict__ U2,
                                          int n, int kk) {
    if (kk < 368) return U3[n * 368 + kk];          // U3[w,v,i,k], kk = i*23+k
    if (kk < KKUSE) return U2[n * 4 + (kk - 368)];  // U2[w,v,k2]
    return 0.0f;                                    // K padding
}

// ---------------------------------------------------------------------------
// Preprocess: pack U3 (+U2 tail) into bf16 WMMA B-fragments.
// Layout: frag(ks, ntile) -> 32 lanes x 8 dwords, stored [ks][ntile][lane][j].
// B layout (32x16 bf16): lanes 0-15 hold Klocal 0..7,16..23 (dword j: K=16*(j>>2)+2*(j&3),+1),
// lanes 16-31 hold Klocal 8..15,24..31 (same, +8); column N = ntile*16 + (lane & 15).
// ---------------------------------------------------------------------------
__global__ __launch_bounds__(256) void build_u3b_kernel(
    const float* __restrict__ U3, const float* __restrict__ U2,
    unsigned int* __restrict__ U3B) {
    int tid   = blockIdx.x * 256 + threadIdx.x;   // 0 .. 49151
    int j     = tid & 7;
    int lane  = (tid >> 3) & 31;
    int ntile = (tid >> 8) & 15;
    int ks    = tid >> 12;
    int n     = ntile * 16 + (lane & 15);
    int kk0   = ks * 32 + ((lane & 16) ? 8 : 0) + ((j & 4) ? 16 : 0) + 2 * (j & 3);
    unsigned int lo = f2bf(u3t_elem(U3, U2, n, kk0));
    unsigned int hi = f2bf(u3t_elem(U3, U2, n, kk0 + 1));
    U3B[tid] = lo | (hi << 16);
}

// ---------------------------------------------------------------------------
// Main fused kernel: one workgroup = 64 c-rows of one b, 2 waves.
//   phase 1: stage x tile (64x16 f32) into LDS
//   phase 2: build Q (64x384 bf16) in LDS: Q[c][i*23+k] = x[c][i]*w3[k],
//            Q[c][368+k2] = w2[k2], zero pad
//   phase 3: per-wave 32x256x384 bf16 WMMA GEMM (2 M-tiles share each B
//            fragment), fused bilinear epilogue
// ---------------------------------------------------------------------------
__global__ __launch_bounds__(NTHR) void symcon_kernel(
    const float* __restrict__ x,  const float* __restrict__ y,
    const float* __restrict__ U1,
    const float* __restrict__ W3, const float* __restrict__ W2,
    const float* __restrict__ W1,
    const unsigned int* __restrict__ U3B,
    float* __restrict__ out) {

    __shared__ __align__(16) unsigned short Qlds[RM * KKPAD];  // 48 KB
    __shared__ __align__(16) float          xlds[RM * LN];     // 4 KB

    const int tid = threadIdx.x;
    const int b   = blockIdx.x >> 2;
    const int c0  = (blockIdx.x & 3) * RM;

    // ---- phase 1: stage x tile (1024 floats, 64 threads x 4 float4) ----
    {
        const float4* xs = reinterpret_cast<const float4*>(
            x + (size_t)(b * CN + c0) * LN);
        float4* xd = reinterpret_cast<float4*>(xlds);
#pragma unroll
        for (int q = 0; q < 4; ++q) xd[tid * 4 + q] = xs[tid * 4 + q];
    }
    __syncthreads();

    // ---- phase 2: build Q (one thread per row) ----
    {
        const int rc = tid;
        const int c  = c0 + rc;
        float w3r[K3N];
#pragma unroll
        for (int k = 0; k < K3N; ++k) w3r[k] = 0.0f;
        for (int e = 0; e < EN; ++e) {
            float ye = y[b * EN + e];
#pragma unroll
            for (int k = 0; k < K3N; ++k)
                w3r[k] += W3[(e * K3N + k) * CN + c] * ye;
        }
        unsigned short* qrow = Qlds + rc * KKPAD;
#pragma unroll
        for (int i = 0; i < LN; ++i) {
            float xi = xlds[rc * LN + i];
#pragma unroll
            for (int k = 0; k < K3N; ++k)
                qrow[i * K3N + k] = f2bf(xi * w3r[k]);
        }
        for (int k2 = 0; k2 < K2N; ++k2) {
            float w2v = 0.0f;
            for (int e = 0; e < EN; ++e)
                w2v += W2[(e * K2N + k2) * CN + c] * y[b * EN + e];
            qrow[368 + k2] = f2bf(w2v);
        }
        for (int kk = KKUSE; kk < KKPAD; ++kk) qrow[kk] = 0;
    }
    __syncthreads();

    // ---- phase 3: WMMA GEMM + fused epilogue ----
    const int lane = tid & 31;
    const int wave = tid >> 5;          // 0..1; wave owns M-tiles 2w, 2w+1
    const int hf   = lane >> 4;         // half-wave select

    // A fragment (16x32 bf16): lane row = lane&15; VGPR0-3: K = hf*8+2j,
    // VGPR4-7: K = 16 + hf*8 + 2j.  As dwords in Qlds row: two uint4 loads.
    const uint4* Qv = reinterpret_cast<const uint4*>(Qlds);
    const uint4* Bv = reinterpret_cast<const uint4*>(U3B);
    const int arow0  = wave * 32 + (lane & 15);
    const int abase0 = arow0 * 48 + hf;          // uint4 index; + ks*4 (+2)
    const int abase1 = abase0 + 16 * 48;         // second M-tile, +16 rows
    const int bbase  = lane * 2;                 // uint4 index; + (ks*16+nt)*64

    float acc[2][8];
#pragma unroll
    for (int m = 0; m < 2; ++m)
#pragma unroll
        for (int r = 0; r < 8; ++r) acc[m][r] = 0.0f;

    for (int ng = 0; ng < 4; ++ng) {          // groups of 4 N-tiles
        v8f D[2][4] = {};
        for (int ks = 0; ks < NKS; ++ks) {
            uint4 a0 = Qv[abase0 + ks * 4];
            uint4 a1 = Qv[abase0 + ks * 4 + 2];
            v8u   au0 = {a0.x, a0.y, a0.z, a0.w, a1.x, a1.y, a1.z, a1.w};
            v16bf A0  = __builtin_bit_cast(v16bf, au0);
            uint4 a2 = Qv[abase1 + ks * 4];
            uint4 a3 = Qv[abase1 + ks * 4 + 2];
            v8u   au1 = {a2.x, a2.y, a2.z, a2.w, a3.x, a3.y, a3.z, a3.w};
            v16bf A1  = __builtin_bit_cast(v16bf, au1);
#pragma unroll
            for (int t = 0; t < 4; ++t) {
                int   nt = ng * 4 + t;
                int   bi = bbase + (ks * 16 + nt) * 64;
                uint4 b0 = Bv[bi];
                uint4 b1 = Bv[bi + 1];
                v8u   bu = {b0.x, b0.y, b0.z, b0.w, b1.x, b1.y, b1.z, b1.w};
                v16bf Bf = __builtin_bit_cast(v16bf, bu);
                D[0][t] = __builtin_amdgcn_wmma_f32_16x16x32_bf16(
                    false, A0, false, Bf, (short)0, D[0][t], false, false);
                D[1][t] = __builtin_amdgcn_wmma_f32_16x16x32_bf16(
                    false, A1, false, Bf, (short)0, D[1][t], false, false);
            }
        }
        // fused bilinear epilogue: acc += c2[c, w*16+v] * x[c,v] * x[c,w]
        // D VGPR r, lane l: row M = r + 8*(l>>4), col v = l&15; N-tile == w.
#pragma unroll
        for (int m = 0; m < 2; ++m) {
#pragma unroll
            for (int t = 0; t < 4; ++t) {
                int w = ng * 4 + t;
#pragma unroll
                for (int r = 0; r < 8; ++r) {
                    int   crow = wave * 32 + m * 16 + r + 8 * hf;
                    float xv   = xlds[crow * LN + (lane & 15)];
                    float xw   = xlds[crow * LN + w];
                    acc[m][r] += D[m][t][r] * xv * xw;
                }
            }
        }
    }

    // reduce across the 16 columns (lanes sharing a row) and write out,
    // adding the nu=1 correction  w1[b,c] * sum_i U1[i]*x[c,i].
#pragma unroll
    for (int m = 0; m < 2; ++m) {
#pragma unroll
        for (int r = 0; r < 8; ++r) {
            float red = acc[m][r];
            red += __shfl_xor(red, 1, 32);
            red += __shfl_xor(red, 2, 32);
            red += __shfl_xor(red, 4, 32);
            red += __shfl_xor(red, 8, 32);
            if ((lane & 15) == 0) {
                int   crow = wave * 32 + m * 16 + r + 8 * hf;
                int   c    = c0 + crow;
                float w1c = 0.0f, u1x = 0.0f;
                for (int e = 0; e < EN; ++e)
                    w1c += W1[e * CN + c] * y[b * EN + e];
#pragma unroll
                for (int i = 0; i < LN; ++i)
                    u1x += U1[i] * xlds[crow * LN + i];
                out[b * CN + c] = red + w1c * u1x;
            }
        }
    }
}

extern "C" void kernel_launch(void* const* d_in, const int* in_sizes, int n_in,
                              void* d_out, int out_size, void* d_ws, size_t ws_size,
                              hipStream_t stream) {
    const float* x  = (const float*)d_in[0];
    const float* y  = (const float*)d_in[1];
    const float* U3 = (const float*)d_in[2];
    const float* U2 = (const float*)d_in[3];
    const float* U1 = (const float*)d_in[4];
    const float* W3 = (const float*)d_in[5];
    const float* W2 = (const float*)d_in[6];
    const float* W1 = (const float*)d_in[7];
    float* out = (float*)d_out;

    // scratch: bf16 B-fragments, 12*16*32*8 dwords = 192 KB
    unsigned int* U3B = (unsigned int*)d_ws;

    build_u3b_kernel<<<192, 256, 0, stream>>>(U3, U2, U3B);
    symcon_kernel<<<BN * 4, NTHR, 0, stream>>>(x, y, U1, W3, W2, W1, U3B, out);
}
